// GraphAttention_53231824666769
// MI455X (gfx1250) — compile-verified, matrix operand order checked by reading, not run on hardware
//
#include <hip/hip_runtime.h>

typedef __attribute__((ext_vector_type(16))) __bf16 v16bf;
typedef __attribute__((ext_vector_type(8)))  float  v8f;
typedef __attribute__((ext_vector_type(4)))  unsigned u32x4;
typedef __attribute__((ext_vector_type(8)))  int      i32x8;
typedef __attribute__((ext_vector_type(4)))  int      i32x4;

#define SCALE_ 0.125f  // HD^-0.5, HD=64

#if __has_builtin(__builtin_amdgcn_tensor_load_to_lds)
#define HAVE_TDM 1
#else
#define HAVE_TDM 0
#endif

static __device__ __forceinline__ unsigned short f2bf(float x) {
  unsigned int u = __float_as_uint(x);
  u += 0x7fffu + ((u >> 16) & 1u);          // round-to-nearest-even
  return (unsigned short)(u >> 16);
}

static __device__ __forceinline__ v8f wmma_bf16(v16bf a, v16bf b, v8f c) {
  return __builtin_amdgcn_wmma_f32_16x16x32_bf16(false, a, false, b, (short)0, c,
                                                 false, false);
}

static __device__ __forceinline__ v8f vzero8() {
  v8f v;
#pragma unroll
  for (int i = 0; i < 8; ++i) v[i] = 0.0f;
  return v;
}

#if HAVE_TDM
// TDM: copy a 128(rows) x 32(elem) bf16 tile (row pitch 1024 elems) into LDS,
// padding 16B after every 64B row -> LDS row stride 40 elems, matching the
// WMMA fragment loads. Descriptor per CDNA5 ISA ch.8 (group0/group1).
static __device__ __forceinline__ void tdm_load_tile(const unsigned short* gsrc,
                                                     unsigned lds_byte_off,
                                                     unsigned tensor_rows) {
  unsigned long long ga = (unsigned long long)(uintptr_t)gsrc;
  u32x4 g0;
  g0[0] = 1u;                                   // count=1, user descriptor
  g0[1] = lds_byte_off;                         // lds_addr
  g0[2] = (unsigned)ga;                         // global_addr[31:0]
  g0[3] = (unsigned)((ga >> 32) & 0x01ffffffu)  // global_addr[56:32]
          | (2u << 30);                         // type=2 (image)
  i32x8 g1;
  g1[0] = (1 << 16)      // data_size = 2 bytes
        | (1 << 20)      // pad_enable
        | (3 << 22)      // pad_interval: 16 DWORDs (=64B = one 32-elem row)
        | (3 << 25);     // pad_amount: 4 DWORDs (=16B = 8 elems) -> stride 40
  g1[1] = (int)(1024u << 16);          // tensor_dim0 = 1024 (bits 79:48 low part)
  g1[2] = (int)(tensor_rows << 16);    // tensor_dim1 low 16 bits
  g1[3] = (int)(32u << 16);            // tile_dim0 = 32 elems
  g1[4] = 128;                         // tile_dim1 = 128 rows, tile_dim2 = 0
  g1[5] = 1024;                        // tensor_dim0_stride = 1024 elems
  g1[6] = 0;
  g1[7] = 0;
  i32x4 z4;
  z4[0] = z4[1] = z4[2] = z4[3] = 0;
#if defined(__clang_major__) && __clang_major__ >= 23
  i32x8 z8;
#pragma unroll
  for (int i = 0; i < 8; ++i) z8[i] = 0;
  __builtin_amdgcn_tensor_load_to_lds(g0, g1, z4, z4, z8, 0);
#else
  __builtin_amdgcn_tensor_load_to_lds(g0, g1, z4, z4, 0);
#endif
}
#endif  // HAVE_TDM

__global__ void cvt_f32_bf16(const float* __restrict__ src,
                             unsigned short* __restrict__ dst, int n) {
  int i = blockIdx.x * 256 + threadIdx.x;
  if (i < n) dst[i] = f2bf(src[i]);
}

// One-time 1024x1024 f32->bf16 transpose: dst[n][k] = src[k][n].
__global__ __launch_bounds__(256) void cvt_transpose_bf16(
    const float* __restrict__ src, unsigned short* __restrict__ dst) {
  __shared__ float tile[32][33];
  const int bx = blockIdx.x * 32, by = blockIdx.y * 32;
  const int tx = threadIdx.x & 31, ty = threadIdx.x >> 5;  // 32x8
#pragma unroll
  for (int i = 0; i < 32; i += 8)
    tile[ty + i][tx] = src[(size_t)(by + ty + i) * 1024 + bx + tx];
  __syncthreads();
#pragma unroll
  for (int i = 0; i < 32; i += 8)
    dst[(size_t)(bx + ty + i) * 1024 + by + tx] = f2bf(tile[tx][ty + i]);
}

// C[M x 1024] = A[M x 1024](bf16) @ Wt^T + bias, where Wt is [n][k] bf16.
// 128x128x32 tiles, TDM double-buffered LDS staging.
// mode 0: dst bf16 [b,h,s,d]   (Q/K head-split)
// mode 1: dst bf16 [b,h,d,s]   (V transposed for P@V B-operand)
// mode 2: dst f32  [m,n]       (final output projection)
__global__ __launch_bounds__(256) void gemm_bf16(
    const unsigned short* __restrict__ A, const unsigned short* __restrict__ Wt,
    const float* __restrict__ bias, void* __restrict__ dst, int mode) {
  __shared__ unsigned short lds_a[2][128 * 40];  // stride 40: 32 data + 8 pad
  __shared__ unsigned short lds_b[2][128 * 40];  // stored [n][k]

  const int tid   = threadIdx.x;
  const int lane  = tid & 31;
  const int wave  = tid >> 5;
  const int wm    = wave >> 2;   // 0..1  (64 rows each)
  const int wn    = wave & 3;    // 0..3  (32 cols each)
  const int m0    = blockIdx.y * 128;
  const int n0    = blockIdx.x * 128;
  const int half8 = (lane & 16) ? 8 : 0;   // 16-bit A/B K-split across half-wave
  const int l16   = lane & 15;

  v8f acc[4][2];
#pragma unroll
  for (int i = 0; i < 4; ++i)
#pragma unroll
    for (int j = 0; j < 2; ++j) acc[i][j] = vzero8();

#if HAVE_TDM
  const unsigned a_off[2] = {(unsigned)(uintptr_t)&lds_a[0][0],
                             (unsigned)(uintptr_t)&lds_a[1][0]};
  const unsigned b_off[2] = {(unsigned)(uintptr_t)&lds_b[0][0],
                             (unsigned)(uintptr_t)&lds_b[1][0]};
  if (wave == 0) {
    tdm_load_tile(A + (size_t)m0 * 1024, a_off[0], 4096u);
    tdm_load_tile(Wt + (size_t)n0 * 1024, b_off[0], 1024u);
  }
#endif

  for (int it = 0; it < 32; ++it) {
    const int p = it & 1;
    const int kk = it * 32;
#if HAVE_TDM
    if (wave == 0) {
      if (it + 1 < 32) {  // prefetch next stage into other buffer via TDM
        tdm_load_tile(A + (size_t)m0 * 1024 + (kk + 32), a_off[p ^ 1], 4096u);
        tdm_load_tile(Wt + (size_t)n0 * 1024 + (kk + 32), b_off[p ^ 1], 1024u);
        __builtin_amdgcn_s_wait_tensorcnt((short)2);  // current stage complete
      } else {
        __builtin_amdgcn_s_wait_tensorcnt((short)0);
      }
    }
    __syncthreads();
#else
    {  // fallback manual staging (single logical buffer per parity)
      const int arow_ld = tid >> 1;
      const int aseg = tid & 1;
      const uint4* srcp =
          (const uint4*)(A + (size_t)(m0 + arow_ld) * 1024 + kk + aseg * 16);
      uint4* d = (uint4*)(&lds_a[p][arow_ld * 40 + aseg * 16]);
      d[0] = srcp[0];
      d[1] = srcp[1];
      const uint4* srcw =
          (const uint4*)(Wt + (size_t)(n0 + arow_ld) * 1024 + kk + aseg * 16);
      uint4* db = (uint4*)(&lds_b[p][arow_ld * 40 + aseg * 16]);
      db[0] = srcw[0];
      db[1] = srcw[1];
    }
    __syncthreads();
#endif

    v16bf fa[4], fb[2];
#pragma unroll
    for (int ms = 0; ms < 4; ++ms) {
      const unsigned short* pa = &lds_a[p][(wm * 64 + ms * 16 + l16) * 40 + half8];
      ((uint4*)&fa[ms])[0] = *(const uint4*)(pa);
      ((uint4*)&fa[ms])[1] = *(const uint4*)(pa + 16);
    }
#pragma unroll
    for (int ns = 0; ns < 2; ++ns) {
      const unsigned short* pb = &lds_b[p][(wn * 32 + ns * 16 + l16) * 40 + half8];
      ((uint4*)&fb[ns])[0] = *(const uint4*)(pb);
      ((uint4*)&fb[ns])[1] = *(const uint4*)(pb + 16);
    }
#pragma unroll
    for (int ms = 0; ms < 4; ++ms)
#pragma unroll
      for (int ns = 0; ns < 2; ++ns)
        acc[ms][ns] = wmma_bf16(fa[ms], fb[ns], acc[ms][ns]);
    __syncthreads();  // readers done before this buffer is DMA-overwritten
  }

#pragma unroll
  for (int ms = 0; ms < 4; ++ms)
#pragma unroll
    for (int ns = 0; ns < 2; ++ns)
#pragma unroll
      for (int r = 0; r < 8; ++r) {
        int row = m0 + wm * 64 + ms * 16 + r + half8;
        int col = n0 + wn * 32 + ns * 16 + l16;
        float v = acc[ms][ns][r] + bias[col];
        if (mode == 0) {
          int b = row >> 10, s = row & 1023, h = col >> 6, d = col & 63;
          ((unsigned short*)dst)[(((size_t)(b * 16 + h)) * 1024 + s) * 64 + d] = f2bf(v);
        } else if (mode == 1) {
          int b = row >> 10, s = row & 1023, h = col >> 6, d = col & 63;
          ((unsigned short*)dst)[(((size_t)(b * 16 + h)) * 64 + d) * 1024 + s] = f2bf(v);
        } else {
          ((float*)dst)[(size_t)row * 1024 + col] = v;
        }
      }
}

// One block per (b, h, 16 q-rows). 8 waves each own 128 k-columns.
// Scores stay in accumulators; dual softmax via shuffle+LDS; P kept bf16 in LDS
// so the P@V GEMM never re-reads the 256MB attn_weights from HBM.
__global__ __launch_bounds__(256) void attn_dual(
    const unsigned short* __restrict__ Qb, const unsigned short* __restrict__ Kb,
    const unsigned short* __restrict__ Vt, const unsigned short* __restrict__ Gb,
    float* __restrict__ attnw, unsigned short* __restrict__ Ab) {
  __shared__ float lds_red[8 * 16];
  __shared__ float lds_o[16 * 64];
  __shared__ unsigned short lds_p[16 * 1032];  // padded stride 1032

  const int tid   = threadIdx.x;
  const int lane  = tid & 31;
  const int wave  = tid >> 5;
  const int l16   = lane & 15;
  const int half8 = (lane & 16) ? 8 : 0;
  const int b  = blockIdx.z;
  const int h  = blockIdx.y;
  const int q0 = blockIdx.x * 16;

  for (int i = tid; i < 1024; i += 256) lds_o[i] = 0.0f;

  const size_t bh      = (size_t)(b * 16 + h);
  const size_t qk_base = bh * 1024 * 64;

  v16bf fq[2], fg[2];
  {
    int q = q0 + l16;
    const unsigned short* pq = Qb + qk_base + (size_t)q * 64;
    const unsigned short* pg = Gb + ((size_t)b * 1024 + q) * 64;
#pragma unroll
    for (int c = 0; c < 2; ++c) {
      ((uint4*)&fq[c])[0] = *(const uint4*)(pq + c * 32 + half8);
      ((uint4*)&fq[c])[1] = *(const uint4*)(pq + c * 32 + half8 + 16);
      ((uint4*)&fg[c])[0] = *(const uint4*)(pg + c * 32 + half8);
      ((uint4*)&fg[c])[1] = *(const uint4*)(pg + c * 32 + half8 + 16);
    }
  }

  v8f sq[8], sg[8];
#pragma unroll
  for (int kc = 0; kc < 8; ++kc) { sq[kc] = vzero8(); sg[kc] = vzero8(); }

#pragma unroll
  for (int kc = 0; kc < 8; ++kc) {
    int scol = wave * 128 + kc * 16 + l16;
    const unsigned short* pk = Kb + qk_base + (size_t)scol * 64;
    v16bf fk[2];
#pragma unroll
    for (int c = 0; c < 2; ++c) {
      ((uint4*)&fk[c])[0] = *(const uint4*)(pk + c * 32 + half8);
      ((uint4*)&fk[c])[1] = *(const uint4*)(pk + c * 32 + half8 + 16);
    }
    sq[kc] = wmma_bf16(fq[0], fk[0], sq[kc]);
    sq[kc] = wmma_bf16(fq[1], fk[1], sq[kc]);
    sg[kc] = wmma_bf16(fg[0], fk[0], sg[kc]);
    sg[kc] = wmma_bf16(fg[1], fk[1], sg[kc]);
  }

  float Mq[8], Mg[8], Zq[8], Zg[8];

  // --- row max, seq (scaled) ---
#pragma unroll
  for (int r = 0; r < 8; ++r) {
    float m = -3.0e38f;
#pragma unroll
    for (int kc = 0; kc < 8; ++kc) m = fmaxf(m, sq[kc][r]);
    m *= SCALE_;
#pragma unroll
    for (int off = 1; off < 16; off <<= 1) m = fmaxf(m, __shfl_xor(m, off, 16));
    Mq[r] = m;
  }
  if (l16 == 0)
#pragma unroll
    for (int r = 0; r < 8; ++r) lds_red[wave * 16 + r + half8] = Mq[r];
  __syncthreads();
#pragma unroll
  for (int r = 0; r < 8; ++r) {
    float m = -3.0e38f;
#pragma unroll
    for (int w = 0; w < 8; ++w) m = fmaxf(m, lds_red[w * 16 + r + half8]);
    Mq[r] = m;
  }
  __syncthreads();

  // --- row max, graph ---
#pragma unroll
  for (int r = 0; r < 8; ++r) {
    float m = -3.0e38f;
#pragma unroll
    for (int kc = 0; kc < 8; ++kc) m = fmaxf(m, sg[kc][r]);
#pragma unroll
    for (int off = 1; off < 16; off <<= 1) m = fmaxf(m, __shfl_xor(m, off, 16));
    Mg[r] = m;
  }
  if (l16 == 0)
#pragma unroll
    for (int r = 0; r < 8; ++r) lds_red[wave * 16 + r + half8] = Mg[r];
  __syncthreads();
#pragma unroll
  for (int r = 0; r < 8; ++r) {
    float m = -3.0e38f;
#pragma unroll
    for (int w = 0; w < 8; ++w) m = fmaxf(m, lds_red[w * 16 + r + half8]);
    Mg[r] = m;
  }
  __syncthreads();

  // --- exp & row sums (overwrite score regs with exp values) ---
#pragma unroll
  for (int r = 0; r < 8; ++r) {
    float s = 0.0f;
#pragma unroll
    for (int kc = 0; kc < 8; ++kc) {
      float e = __expf(sq[kc][r] * SCALE_ - Mq[r]);
      sq[kc][r] = e;
      s += e;
    }
#pragma unroll
    for (int off = 1; off < 16; off <<= 1) s += __shfl_xor(s, off, 16);
    Zq[r] = s;
  }
  if (l16 == 0)
#pragma unroll
    for (int r = 0; r < 8; ++r) lds_red[wave * 16 + r + half8] = Zq[r];
  __syncthreads();
#pragma unroll
  for (int r = 0; r < 8; ++r) {
    float s = 0.0f;
#pragma unroll
    for (int w = 0; w < 8; ++w) s += lds_red[w * 16 + r + half8];
    Zq[r] = s;
  }
  __syncthreads();

#pragma unroll
  for (int r = 0; r < 8; ++r) {
    float s = 0.0f;
#pragma unroll
    for (int kc = 0; kc < 8; ++kc) {
      float e = __expf(sg[kc][r] - Mg[r]);
      sg[kc][r] = e;
      s += e;
    }
#pragma unroll
    for (int off = 1; off < 16; off <<= 1) s += __shfl_xor(s, off, 16);
    Zg[r] = s;
  }
  if (l16 == 0)
#pragma unroll
    for (int r = 0; r < 8; ++r) lds_red[wave * 16 + r + half8] = Zg[r];
  __syncthreads();
#pragma unroll
  for (int r = 0; r < 8; ++r) {
    float s = 0.0f;
#pragma unroll
    for (int w = 0; w < 8; ++w) s += lds_red[w * 16 + r + half8];
    Zg[r] = s;
  }
  __syncthreads();

  // --- fused dual-softmax weights: write f32 to d_out, bf16 tile to LDS ---
#pragma unroll
  for (int r = 0; r < 8; ++r) {
    int row = r + half8;
    float inv = 1.0f / (Zq[r] * Zg[r]);
    size_t obase = (bh * 1024 + (size_t)(q0 + row)) * 1024;
#pragma unroll
    for (int kc = 0; kc < 8; ++kc) {
      int col = wave * 128 + kc * 16 + l16;
      float w = sq[kc][r] * sg[kc][r] * inv;
      attnw[obase + col] = w;
      lds_p[row * 1032 + col] = f2bf(w);
    }
  }
  __syncthreads();

  // --- P(16x1024, LDS) @ V(1024x64): wave w covers its 128 k-columns ---
  v8f oacc[4];
#pragma unroll
  for (int dt = 0; dt < 4; ++dt) oacc[dt] = vzero8();
#pragma unroll
  for (int kc2 = 0; kc2 < 4; ++kc2) {
    int sbase = wave * 128 + kc2 * 32;
    const unsigned short* pp = lds_p + l16 * 1032 + sbase;
    v16bf fp;
    ((uint4*)&fp)[0] = *(const uint4*)(pp + half8);
    ((uint4*)&fp)[1] = *(const uint4*)(pp + half8 + 16);
#pragma unroll
    for (int dt = 0; dt < 4; ++dt) {
      int dcol = dt * 16 + l16;
      const unsigned short* pv = Vt + (bh * 64 + dcol) * 1024 + sbase;
      v16bf fv;
      ((uint4*)&fv)[0] = *(const uint4*)(pv + half8);
      ((uint4*)&fv)[1] = *(const uint4*)(pv + half8 + 16);
      oacc[dt] = wmma_bf16(fp, fv, oacc[dt]);
    }
  }
#pragma unroll
  for (int dt = 0; dt < 4; ++dt)
#pragma unroll
    for (int r = 0; r < 8; ++r)
      atomicAdd(&lds_o[(r + half8) * 64 + dt * 16 + l16], oacc[dt][r]);  // ds_add_f32
  __syncthreads();

  for (int i = tid; i < 1024; i += 256) {
    int row = i >> 6, d = i & 63;
    Ab[((size_t)b * 1024 + (q0 + row)) * 1024 + h * 64 + d] = f2bf(lds_o[i]);
  }
}

extern "C" void kernel_launch(void* const* d_in, const int* in_sizes, int n_in,
                              void* d_out, int out_size, void* d_ws, size_t ws_size,
                              hipStream_t stream) {
  const float* X  = (const float*)d_in[0];
  const float* G  = (const float*)d_in[1];
  const float* Wq = (const float*)d_in[2];
  const float* bq = (const float*)d_in[3];
  const float* Wk = (const float*)d_in[4];
  const float* bk = (const float*)d_in[5];
  const float* Wv = (const float*)d_in[6];
  const float* bv = (const float*)d_in[7];
  const float* Wo = (const float*)d_in[8];
  const float* bo = (const float*)d_in[9];

  unsigned short* ws = (unsigned short*)d_ws;
  const size_t M4 = 4194304;  // 4M elements
  unsigned short* Xb  = ws;                       // X bf16        [4096 x 1024]
  unsigned short* Wqt = ws + M4;                  // W^T bf16 [n][k] x4
  unsigned short* Wkt = ws + M4 + 1048576;
  unsigned short* Wvt = ws + M4 + 2097152;
  unsigned short* Wot = ws + M4 + 3145728;
  unsigned short* Qb  = ws + 2 * M4;              // [b,h,s,d]
  unsigned short* Kb  = ws + 3 * M4;              // [b,h,s,d]
  unsigned short* Vt  = ws + 4 * M4;              // [b,h,d,s]
  unsigned short* Gb  = ws + 5 * M4;              // [b,s,d]
  unsigned short* Ab  = ws + 5 * M4 + 262144;     // attn_out bf16 [4096 x 1024]

  float* out   = (float*)d_out;
  float* attnw = out + M4;  // [B,H,S,S] f32

  cvt_f32_bf16<<<dim3(16384), 256, 0, stream>>>(X, Xb, 4194304);
  cvt_f32_bf16<<<dim3(1024), 256, 0, stream>>>(G, Gb, 262144);
  dim3 tg(32, 32);
  cvt_transpose_bf16<<<tg, 256, 0, stream>>>(Wq, Wqt);
  cvt_transpose_bf16<<<tg, 256, 0, stream>>>(Wk, Wkt);
  cvt_transpose_bf16<<<tg, 256, 0, stream>>>(Wv, Wvt);
  cvt_transpose_bf16<<<tg, 256, 0, stream>>>(Wo, Wot);

  dim3 gg(8, 32);  // N tiles x M tiles
  gemm_bf16<<<gg, 256, 0, stream>>>(Xb, Wqt, bq, (void*)Qb, 0);
  gemm_bf16<<<gg, 256, 0, stream>>>(Xb, Wkt, bk, (void*)Kb, 0);
  gemm_bf16<<<gg, 256, 0, stream>>>(Xb, Wvt, bv, (void*)Vt, 1);

  attn_dual<<<dim3(64, 16, 4), 256, 0, stream>>>(Qb, Kb, Vt, Gb, attnw, Ab);

  gemm_bf16<<<gg, 256, 0, stream>>>(Ab, Wot, bo, (void*)out, 2);
}
